// AttentionNCF_67061619360220
// MI455X (gfx1250) — compile-verified, matrix operand order checked by reading, not run on hardware
//
#include <hip/hip_runtime.h>

typedef __attribute__((ext_vector_type(16))) _Float16 v16h;
typedef __attribute__((ext_vector_type(8)))  _Float16 v8h;
typedef __attribute__((ext_vector_type(8)))  float    v8f;

#define KP 1024   // K padded (D = 1000 -> 1024), zero filled

// ---------------------------------------------------------------------------
// f32 -> f16 convert with zero padding to [prows][pcols]
// ---------------------------------------------------------------------------
__global__ void cvt_pad_f16(const float* __restrict__ src, _Float16* __restrict__ dst,
                            int rows, int cols, int prows, int pcols) {
  long idx = (long)blockIdx.x * blockDim.x + threadIdx.x;
  long total = (long)prows * pcols;
  if (idx >= total) return;
  int r = (int)(idx / pcols);
  int c = (int)(idx % pcols);
  float v = (r < rows && c < cols) ? src[(long)r * cols + c] : 0.0f;
  dst[idx] = (_Float16)v;
}

// Build the 16-bit WMMA A/B operand for one lane.
// Lanes 0-15 carry K {kb..kb+7, kb+16..kb+23}; lanes 16-31 carry {kb+8..15, kb+24..31}.
__device__ __forceinline__ v16h load_op16(const _Float16* base, int kb, bool hi_half) {
  const int o = kb + (hi_half ? 8 : 0);
  v8h lo = *(const v8h*)(base + o);
  v8h hi = *(const v8h*)(base + o + 16);
  v16h r;
#pragma unroll
  for (int j = 0; j < 8; ++j) { r[j] = lo[j]; r[j + 8] = hi[j]; }
  return r;
}

// ---------------------------------------------------------------------------
// Embedding GEMM:  out[M][64] = A[M][KP](f16) @ We^T + be
// B operand column n == webf row n (We is (E,D) row-major => contiguous in K).
// Optionally also writes ert[64][KP] = out^T in f16 (zero for rows >= validRows),
// which becomes the B operand of the fused user_emb GEMM.
// One wave computes one 16x16 tile; 8 waves / block.
// ---------------------------------------------------------------------------
__global__ void __launch_bounds__(256)
gemm_embed(const _Float16* __restrict__ A,
           const _Float16* __restrict__ webf,   // [64][KP]
           const float* __restrict__ be,        // [64]
           float* __restrict__ out,             // [M][64]
           _Float16* __restrict__ ert,          // [64][KP] or nullptr
           int validRows) {
  const int lane  = threadIdx.x & 31;
  const int wid   = threadIdx.x >> 5;
  const int tile  = blockIdx.x * 8 + wid;
  const int mtile = tile >> 2;
  const int ntile = tile & 3;
  const bool hi   = lane >= 16;

  const _Float16* arow = A    + (long)(mtile * 16 + (lane & 15)) * KP;
  const _Float16* bcol = webf + (long)(ntile * 16 + (lane & 15)) * KP;

  v8f acc = {};
#pragma unroll 8
  for (int kb = 0; kb < KP; kb += 32) {
    v16h a = load_op16(arow, kb, hi);
    v16h b = load_op16(bcol, kb, hi);
    acc = __builtin_amdgcn_wmma_f32_16x16x32_f16(false, a, false, b,
                                                 (short)0, acc, false, false);
  }

  // D layout: element r -> (m = r + (hi?8:0), n = lane&15)
  const int n  = ntile * 16 + (lane & 15);
  const int mb = mtile * 16 + (hi ? 8 : 0);
  const float bias = be[n];
#pragma unroll
  for (int r = 0; r < 8; ++r) {
    const int m = mb + r;
    const float v = acc[r] + bias;
    out[(long)m * 64 + n] = v;
    if (ert) ert[(long)n * KP + m] = (m < validRows) ? (_Float16)v : (_Float16)0.0f;
  }
}

// ---------------------------------------------------------------------------
// pr[i][a] = ba1[a] + sum_e e_r[i][e] * Wa1[a][64+e]     (ba1 folded in)
// pc[b][a] =          sum_e e_c[b][e] * Wa1[a][e]
// ---------------------------------------------------------------------------
__global__ void attn_pr(const float* __restrict__ er, const float* __restrict__ Wa1,
                        const float* __restrict__ ba1, float* __restrict__ pr, int I) {
  int i = blockIdx.x * blockDim.x + threadIdx.x;
  if (i >= I) return;
  const float* e = er + (long)i * 64;
#pragma unroll
  for (int a = 0; a < 16; ++a) {
    const float* w = Wa1 + a * 128 + 64;
    float s = ba1[a];
#pragma unroll
    for (int k = 0; k < 64; ++k) s += e[k] * w[k];
    pr[i * 16 + a] = s;
  }
}

__global__ void attn_pc(const float* __restrict__ ec, const float* __restrict__ Wa1,
                        float* __restrict__ pc, int B) {
  int b = blockIdx.x * blockDim.x + threadIdx.x;
  if (b >= B) return;
  const float* e = ec + (long)b * 64;
#pragma unroll
  for (int a = 0; a < 16; ++a) {
    const float* w = Wa1 + a * 128;
    float s = 0.0f;
#pragma unroll
    for (int k = 0; k < 64; ++k) s += e[k] * w[k];
    pc[b * 16 + a] = s;
  }
}

// ---------------------------------------------------------------------------
// Fused: scores -> softmax -> (att*U) -> user_emb WMMA GEMM -> MLP -> out.
// One block handles 16 candidate rows. The 16x1024 score tile lives in LDS
// (f16) and is reused in place as exp values and then as the WMMA A operand.
// ---------------------------------------------------------------------------
__global__ void __launch_bounds__(256)
fused_attention(const float* __restrict__ pc,       // [B][16]
                const float* __restrict__ pr,       // [I][16], ba1 folded in
                const float* __restrict__ Wa2,      // [16]
                const float* __restrict__ U,        // [B][I]
                const _Float16* __restrict__ ert,   // [64][KP]  (e_r^T, f16)
                const float* __restrict__ ec,       // [B][64]
                const float* __restrict__ Wm1, const float* __restrict__ bm1,
                const float* __restrict__ Wm2, const float* __restrict__ bm2,
                const float* __restrict__ Wm3, const float* __restrict__ bm3,
                float* __restrict__ out, int I) {
  __shared__ _Float16 lds_s[16 * KP];   // 32 KB: scores / exp / att*U (A operand)
  __shared__ float    lds_pc[16 * 16];
  __shared__ float    lds_ue[16 * 64];
  __shared__ float    lds_ec[16 * 64];
  __shared__ float    lds_h1[16 * 64];
  __shared__ float    lds_h2[16 * 32];

  const int t    = threadIdx.x;
  const int lane = t & 31;
  const int wid  = t >> 5;
  const int b0   = blockIdx.x * 16;

  // stage pc tile + attention output weights
  lds_pc[t] = pc[(long)(b0 + (t >> 4)) * 16 + (t & 15)];
  float wa2v[16];
#pragma unroll
  for (int a = 0; a < 16; ++a) wa2v[a] = Wa2[a];
  __syncthreads();

  // ---- scores: s[r][i] = sum_a Wa2[a] * relu(pc[r][a] + pr[i][a]) ----
  // (ba2 is constant over i -> softmax invariant -> dropped)
  for (int i = t; i < I; i += 256) {
    const float4* prp = (const float4*)(pr + (long)i * 16);
    const float4 p0 = prp[0], p1 = prp[1], p2 = prp[2], p3 = prp[3];
    const float pa[16] = {p0.x, p0.y, p0.z, p0.w, p1.x, p1.y, p1.z, p1.w,
                          p2.x, p2.y, p2.z, p2.w, p3.x, p3.y, p3.z, p3.w};
#pragma unroll
    for (int r = 0; r < 16; ++r) {
      float s = 0.0f;
#pragma unroll
      for (int a = 0; a < 16; ++a) {
        const float h = lds_pc[r * 16 + a] + pa[a];
        s += wa2v[a] * (h > 0.0f ? h : 0.0f);
      }
      lds_s[r * KP + i] = (_Float16)s;
    }
  }
  __syncthreads();

  // ---- softmax + multiply by user_matrix; wave w owns rows 2w, 2w+1 ----
#pragma unroll
  for (int rr = 0; rr < 2; ++rr) {
    const int r = wid * 2 + rr;
    _Float16* srow = lds_s + r * KP;
    float m = -1e30f;
    for (int i = lane; i < I; i += 32) m = fmaxf(m, (float)srow[i]);
#pragma unroll
    for (int off = 16; off; off >>= 1) m = fmaxf(m, __shfl_xor(m, off, 32));
    float sum = 0.0f;
    for (int i = lane; i < I; i += 32) {
      const float e = __expf((float)srow[i] - m);
      sum += e;
      srow[i] = (_Float16)e;          // in place: exp values
    }
#pragma unroll
    for (int off = 16; off; off >>= 1) sum += __shfl_xor(sum, off, 32);
    const float inv = 1.0f / sum;
    const float* urow = U + (long)(b0 + r) * I;
    for (int i = lane; i < KP; i += 32) {
      const float w = (i < I) ? (float)srow[i] * inv * urow[i] : 0.0f;
      srow[i] = (_Float16)w;          // in place: WMMA A operand (zero padded)
    }
  }
  __syncthreads();

  // ---- user_emb(16x64) = w(16xKP) @ e_r(KPx64): waves 0-3 WMMA, waves 4-7 stage e_c ----
  if (wid < 4) {
    const bool hi  = lane >= 16;
    const int  n   = wid * 16 + (lane & 15);
    const _Float16* arow = lds_s + (long)(lane & 15) * KP;
    const _Float16* bcol = ert   + (long)n * KP;
    v8f acc = {};
#pragma unroll 8
    for (int kb = 0; kb < KP; kb += 32) {
      v16h a = load_op16(arow, kb, hi);
      v16h b = load_op16(bcol, kb, hi);
      acc = __builtin_amdgcn_wmma_f32_16x16x32_f16(false, a, false, b,
                                                   (short)0, acc, false, false);
    }
    const int mb = hi ? 8 : 0;
#pragma unroll
    for (int r = 0; r < 8; ++r) lds_ue[(mb + r) * 64 + n] = acc[r];
  } else {
    for (int idx = t - 128; idx < 16 * 64; idx += 128)
      lds_ec[idx] = ec[(long)b0 * 64 + idx];
  }
  __syncthreads();

  // ---- MLP: x = [e_c | user_emb] -> relu(128->64) -> relu(64->32) -> 1 ----
  for (int idx = t; idx < 16 * 64; idx += 256) {
    const int r = idx >> 6, j = idx & 63;
    const float* w = Wm1 + j * 128;
    float s = bm1[j];
#pragma unroll
    for (int k = 0; k < 64; ++k) s += lds_ec[r * 64 + k] * w[k];
#pragma unroll
    for (int k = 0; k < 64; ++k) s += lds_ue[r * 64 + k] * w[64 + k];
    lds_h1[idx] = s > 0.0f ? s : 0.0f;
  }
  __syncthreads();
  for (int idx = t; idx < 16 * 32; idx += 256) {
    const int r = idx >> 5, j = idx & 31;
    const float* w = Wm2 + j * 64;
    float s = bm2[j];
#pragma unroll
    for (int k = 0; k < 64; ++k) s += lds_h1[r * 64 + k] * w[k];
    lds_h2[idx] = s > 0.0f ? s : 0.0f;
  }
  __syncthreads();
  if (t < 16) {
    float s = bm3[0];
#pragma unroll
    for (int k = 0; k < 32; ++k) s += lds_h2[t * 32 + k] * Wm3[k];
    out[b0 + t] = s;
  }
}

// ---------------------------------------------------------------------------
extern "C" void kernel_launch(void* const* d_in, const int* in_sizes, int n_in,
                              void* d_out, int out_size, void* d_ws, size_t ws_size,
                              hipStream_t stream) {
  const int B = 8192, I = 1000, D = 1000;

  const float* cand  = (const float*)d_in[0];
  const float* rated = (const float*)d_in[1];
  const float* U     = (const float*)d_in[2];
  const float* We    = (const float*)d_in[3];
  const float* be    = (const float*)d_in[4];
  const float* Wa1   = (const float*)d_in[5];
  const float* ba1   = (const float*)d_in[6];
  const float* Wa2   = (const float*)d_in[7];
  /* ba2 = d_in[8]: softmax-invariant, intentionally unused */
  const float* Wm1   = (const float*)d_in[9];
  const float* bm1   = (const float*)d_in[10];
  const float* Wm2   = (const float*)d_in[11];
  const float* bm2   = (const float*)d_in[12];
  const float* Wm3   = (const float*)d_in[13];
  const float* bm3   = (const float*)d_in[14];

  // ---- workspace layout (~22 MB total) ----
  char* ws = (char*)d_ws;
  size_t off = 0;
  auto alloc = [&](size_t bytes) -> void* {
    void* p = ws + off;
    off = (off + bytes + 255) & ~(size_t)255;
    return p;
  };
  _Float16* candp  = (_Float16*)alloc((size_t)B * KP * 2);      // 16.8 MB
  _Float16* ratedp = (_Float16*)alloc((size_t)KP * KP * 2);     //  2.1 MB
  _Float16* webf   = (_Float16*)alloc((size_t)64 * KP * 2);     //  128 KB
  float*    er     = (float*)   alloc((size_t)KP * 64 * 4);     //  256 KB
  _Float16* ert    = (_Float16*)alloc((size_t)64 * KP * 2);     //  128 KB
  float*    ec     = (float*)   alloc((size_t)B * 64 * 4);      //  2.1 MB
  float*    pr     = (float*)   alloc((size_t)KP * 16 * 4);     //   64 KB
  float*    pc     = (float*)   alloc((size_t)B * 16 * 4);      //  512 KB
  (void)ws_size; (void)in_sizes; (void)n_in; (void)out_size;

  // 1) pad/convert operands to f16 (K padded to 1024 with zeros)
  cvt_pad_f16<<<(B * KP) / 256, 256, 0, stream>>>(cand,  candp,  B,  D, B,  KP);
  cvt_pad_f16<<<(KP * KP) / 256, 256, 0, stream>>>(rated, ratedp, I,  D, KP, KP);
  cvt_pad_f16<<<(64 * KP) / 256, 256, 0, stream>>>(We,    webf,   64, D, 64, KP);

  // 2) embedding GEMMs (WMMA f16 -> f32)
  gemm_embed<<< (KP / 16) * 4 / 8, 256, 0, stream>>>(ratedp, webf, be, er, ert, I);  // e_r + e_r^T f16
  gemm_embed<<< (B  / 16) * 4 / 8, 256, 0, stream>>>(candp,  webf, be, ec, nullptr, 0); // e_c

  // 3) attention projections pr, pc (tiny)
  attn_pr<<<KP / 256, 256, 0, stream>>>(er, Wa1, ba1, pr, I);
  attn_pc<<<B  / 256, 256, 0, stream>>>(ec, Wa1, pc, B);

  // 4) fused scores -> softmax -> user_emb WMMA GEMM -> MLP -> output
  fused_attention<<<B / 16, 256, 0, stream>>>(pc, pr, Wa2, U, ert, ec,
                                              Wm1, bm1, Wm2, bm2, Wm3, bm3,
                                              (float*)d_out, I);
}
